// YOLOimgLayer_57741540327770
// MI455X (gfx1250) — matrix operand order repaired and checked by reading, not compiled, not run on hardware
//
#include <hip/hip_runtime.h>
#include <hip/hip_bf16.h>

typedef __attribute__((ext_vector_type(16))) _Float16 v16h;
typedef __attribute__((ext_vector_type(8)))  float    v8f;

#define N_ANC    9
#define N_CH     85
#define OC       765     // 9*85 real output channels
#define OCP      768     // padded to 48 tiles of 16
#define KDIM     256
#define HWSZ     4096    // 64*64
#define MT_TOTAL 48      // OCP/16
#define KC_TOTAL 8       // KDIM/32

// anchors / stride(16), interleaved (w,h) per anchor
__constant__ float c_anc[2 * N_ANC] = {
    10.f/16.f, 13.f/16.f,  16.f/16.f, 30.f/16.f,  33.f/16.f, 23.f/16.f,
    30.f/16.f, 61.f/16.f,  62.f/16.f, 45.f/16.f,  59.f/16.f,119.f/16.f,
   116.f/16.f, 90.f/16.f, 156.f/16.f,198.f/16.f, 373.f/16.f,326.f/16.f
};

// ---------------------------------------------------------------------------
// Kernel 1: fp32 weights (765x256) -> f16, pre-swizzled into WMMA A-fragment
// order (contiguous k = 16*hi + e, 16-bit A 16x32):
//   lane<16 : element e -> K = e                     (K in {0..15})
//   lane>=16: same + 16                              (K in {16..31})
// Storage: wsA[ ((mt*8 + kc)*32 + lane)*16 + e ]
// ---------------------------------------------------------------------------
__global__ __launch_bounds__(256) void yolo_prep_weights(
    const float* __restrict__ w, _Float16* __restrict__ wsA)
{
    const int t = blockIdx.x * 256 + threadIdx.x;     // 0 .. 196607
    const int e    =  t        & 15;
    const int lane = (t >> 4)  & 31;
    const int kc   = (t >> 9)  &  7;
    const int mt   =  t >> 12;
    const int hi   = lane >> 4;
    const int o    = mt * 16 + (lane & 15);
    const int k    = kc * 32 + e + hi * 16;
    const float v  = (o < OC) ? w[o * KDIM + k] : 0.f;
    wsA[t] = (_Float16)v;
}

__device__ __forceinline__ float sigm(float v) {
    return 1.f / (1.f + __expf(-v));
}

// ---------------------------------------------------------------------------
// Kernel 2: GEMM (f16 WMMA, fp32 accum) + YOLO decode epilogue.
// Grid: x = 64 position-strips (64 pos each), y = batch (16), z = 6 M-groups.
// Block: 256 threads = 8 waves; wave w owns M-tile (z*8+w), 4 N-tiles.
// ---------------------------------------------------------------------------
__global__ __launch_bounds__(256) void yolo_gemm_decode(
    const float*    __restrict__ xin,      // (16,256,64,64)
    const float*    __restrict__ bias765,  // (765,)
    const _Float16* __restrict__ wA,       // pre-swizzled A frags
    float*          __restrict__ out)      // (16,9,64,64,85)
{
    // LDS: x chunk as f16, [pos][k] with 32 halves (64B) per pos row so a
    // B-fragment (lane: 16 consecutive K at fixed pos) is one 32B read.
    __shared__ __align__(32) _Float16 sB[64 * 32];

    const int tid  = threadIdx.x;
    const int wave = tid >> 5;
    const int lane = tid & 31;
    const int hi   = lane >> 4;      // half-wave select
    const int ln   = lane & 15;

    const int strip = blockIdx.x;                 // 0..63
    const int b     = blockIdx.y;                 // 0..15
    const int mt    = blockIdx.z * 8 + wave;      // 0..47
    const int pos0  = strip * 64;

    v8f acc[4] = {};

    // cooperative staging: thread -> (pos sp, base-k kb); 8 k's per thread
    const int sp = tid & 63;
    const int kb = tid >> 6;                      // 0..3
    const float* xbase = xin + (size_t)b * KDIM * HWSZ + (size_t)pos0 + sp;

    for (int kc = 0; kc < KC_TOTAL; ++kc) {
        float xv[8];
#pragma unroll
        for (int i = 0; i < 8; ++i) {
            const int k = kb + i * 4;             // covers 0..31
            xv[i] = xbase[(size_t)(kc * 32 + k) * HWSZ];   // coalesced over sp
        }
        __syncthreads();                          // protect prior frag reads
#pragma unroll
        for (int i = 0; i < 8; ++i) {
            const int k = kb + i * 4;
            sB[sp * 32 + k] = (_Float16)xv[i];
        }
        __syncthreads();

        // A fragment: two contiguous 16B loads per lane (pre-swizzled)
        const v16h aF =
            *(const v16h*)(wA + ((size_t)(mt * KC_TOTAL + kc) * 32 + lane) * 16);

#pragma unroll
        for (int nt = 0; nt < 4; ++nt) {
            // B fragment: lanes 0-15 -> K 0..15, lanes 16-31 -> K 16..31
            const v16h bF = *(const v16h*)(sB + (nt * 16 + ln) * 32 + hi * 16);
            acc[nt] = __builtin_amdgcn_wmma_f32_16x16x32_f16(
                false, aF, false, bF, (short)0, acc[nt], false, false);
        }
    }

    // ---------------- epilogue: bias + YOLO decode + scattered store -------
    float bv[8];
#pragma unroll
    for (int r = 0; r < 8; ++r) {
        const int o = mt * 16 + r + 8 * hi;
        bv[r] = (o < OC) ? bias765[o] : 0.f;
    }

#pragma unroll
    for (int nt = 0; nt < 4; ++nt) {
        const int   hw = pos0 + nt * 16 + ln;     // position within (64,64)
        const float fx = (float)(hw & 63);        // x_shift (W index)
        const float fy = (float)(hw >> 6);        // y_shift (H index)
#pragma unroll
        for (int r = 0; r < 8; ++r) {
            const int o = mt * 16 + r + 8 * hi;   // C/D layout: row r + 8*hi
            if (o < OC) {
                const int   a  = o / 85;
                const int   q  = o - a * 85;
                const float wa = c_anc[2 * a + 0];
                const float ha = c_anc[2 * a + 1];
                const float v  = acc[nt][r] + bv[r];
                float res;
                if (q >= 4)      res = sigm(v);
                else if (q == 0) res = ((sigm(v) - 0.5f) * wa + fx) * 16.f;
                else if (q == 1) res = ((sigm(v) - 0.5f) * ha + fy) * 16.f;
                else if (q == 2) res = __expf(v) * wa * 16.f;
                else             res = __expf(v) * ha * 16.f;
                out[(size_t)((b * N_ANC + a) * HWSZ + hw) * N_CH + q] = res;
            }
        }
    }
}

// ---------------------------------------------------------------------------
extern "C" void kernel_launch(void* const* d_in, const int* in_sizes, int n_in,
                              void* d_out, int out_size, void* d_ws, size_t ws_size,
                              hipStream_t stream)
{
    const float* xin    = (const float*)d_in[0];  // 16*256*64*64
    const float* conv_w = (const float*)d_in[1];  // 765*256
    const float* conv_b = (const float*)d_in[2];  // 765
    // d_in[3] = prior_labels: unused on the inference path (matches reference)

    _Float16* wA  = (_Float16*)d_ws;              // needs 768*256*2 = 384 KB
    float*    out = (float*)d_out;

    // 1) swizzle+convert weights into WMMA A-fragment order (every call;
    //    deterministic, graph-capture safe)
    yolo_prep_weights<<<(OCP * KDIM) / 256, 256, 0, stream>>>(conv_w, wA);

    // 2) fused GEMM + decode
    yolo_gemm_decode<<<dim3(64, 16, 6), dim3(256), 0, stream>>>(
        xin, conv_b, wA, out);
}
